// DiagonalFunc_5755256177331
// MI455X (gfx1250) — matrix operand order repaired and checked by reading, not compiled, not run on hardware
//
#include <hip/hip_runtime.h>
#include <hip/hip_bf16.h>

typedef __attribute__((ext_vector_type(16))) _Float16 v16h;
typedef __attribute__((ext_vector_type(8)))  float    v8f;

#define NN      64
#define KIN     65
#define HID     256
#define BATCH   8192
#define BM      16          // rows per wave (one WMMA M tile)
#define WAVES   8           // waves per block
#define BMBLK   (BM*WAVES)  // 128 rows per block

// ---------------------------------------------------------------------------
// Pack z (8192x64 fp32) into f16 A-fragments matching the 16x32 f16 A layout:
//   lane l holds row M = l%16.
//   lanes 0-15 : VGPR v, half hh -> Klocal = (v<4 ? 2v+hh : 16+2(v-4)+hh)
//   lanes 16-31:                     Klocal = (v<4 ? 8+2v+hh : 24+2(v-4)+hh)
// Fragment index: ((tile*2 + kstep)*32 + lane), 16 f16 each. 512 tiles.
// ---------------------------------------------------------------------------
__global__ __launch_bounds__(256) void pack_zA(const float* __restrict__ z,
                                               _Float16* __restrict__ zA) {
  int idx  = blockIdx.x * 256 + threadIdx.x;      // 512*2*32 = 32768 fragments
  int lane = idx & 31;
  int ks   = (idx >> 5) & 1;
  int tile = idx >> 6;
  int row  = tile * 16 + (lane & 15);
  _Float16* dst = zA + (size_t)idx * 16;
#pragma unroll
  for (int j = 0; j < 16; ++j) {
    int v = j >> 1, hh = j & 1;
    int kl;
    if (lane < 16) kl = (v < 4) ? (2*v + hh)     : (16 + 2*(v-4) + hh);
    else           kl = (v < 4) ? (8 + 2*v + hh) : (24 + 2*(v-4) + hh);
    int k = ks * 32 + kl;                          // 0..63 -> z column
    dst[j] = (_Float16)z[(size_t)row * (KIN-1) + k];
  }
}

// ---------------------------------------------------------------------------
// Pack W1[:,1:65,:] (z part) into f16 B-fragments for 32x16 f16 B:
//   lane l holds column N = l%16; lanes 0-15 carry K=j<8 ? j : 16+(j-8),
//   lanes 16-31 carry K=j<8 ? 8+j : 24+(j-8).
// Fragment index: (((n*16 + t)*2 + kstep)*32 + lane), 16 f16 each.
// ---------------------------------------------------------------------------
__global__ __launch_bounds__(256) void pack_w1B(const float* __restrict__ W1,
                                                _Float16* __restrict__ w1B) {
  int idx  = blockIdx.x * 256 + threadIdx.x;      // 64*16*2*32 = 65536 fragments
  int lane = idx & 31;
  int ks   = (idx >> 5) & 1;
  int t    = (idx >> 6) & 15;
  int n    = idx >> 10;
  int col  = t * 16 + (lane & 15);
  _Float16* dst = w1B + (size_t)idx * 16;
#pragma unroll
  for (int j = 0; j < 16; ++j) {
    int K = ((lane >= 16) ? 8 : 0) + ((j < 8) ? j : (16 + (j - 8)));
    int k = ks * 32 + K;                           // 0..63 -> W1 row (k+1)
    dst[j] = (_Float16)W1[((size_t)n * KIN + (k + 1)) * HID + col];
  }
}

// ---------------------------------------------------------------------------
// Main fused kernel: per wave, one 16-row batch tile of one expert.
//   h(16x256) = zA(16x64) @ w1B(64x256)  via v_wmma_f32_16x16x32_f16,
//   + x[:,n]*W1[n,0,:] + b1[n]  (fp32), ELU, dot with W2[n], + b2[n].
// ---------------------------------------------------------------------------
__global__ __launch_bounds__(256) void mlp_fused(
    const float* __restrict__ x,  const float* __restrict__ W1,
    const float* __restrict__ b1, const float* __restrict__ W2,
    const float* __restrict__ b2, const _Float16* __restrict__ zA,
    const _Float16* __restrict__ w1B, float* __restrict__ out) {
  const int lane = threadIdx.x & 31;
  const int wave = threadIdx.x >> 5;
  const int n    = blockIdx.y;
  const int row0 = blockIdx.x * BMBLK + wave * BM;
  const int col  = lane & 15;
  const int half = (lane >> 4) & 1;                // 0: rows v, 1: rows 8+v

  // A fragments for this 16-row tile (reused across all 16 hidden tiles)
  const v16h* zAv  = (const v16h*)zA;
  const v16h* w1Bv = (const v16h*)w1B;
  const int atile  = (row0 >> 4) * 2;
  v16h a0 = zAv[(size_t)(atile + 0) * 32 + lane];
  v16h a1 = zAv[(size_t)(atile + 1) * 32 + lane];

  // x values for the 8 rows this lane's accumulator slots cover (fp32)
  float xs[8];
  const int rbase = row0 + half * 8;
#pragma unroll
  for (int v = 0; v < 8; ++v)
    xs[v] = x[(size_t)(rbase + v) * NN + n];

  float s[8] = {0.f,0.f,0.f,0.f,0.f,0.f,0.f,0.f};

#pragma unroll 4
  for (int t = 0; t < 16; ++t) {
    const size_t bidx = ((size_t)(n * 16 + t) * 2) * 32 + lane;
    v16h bf0 = w1Bv[bidx];
    v16h bf1 = w1Bv[bidx + 32];
    if (t < 15)  // global_prefetch_b8 for next tile's B fragments
      __builtin_prefetch((const void*)(w1B + (bidx + 64) * 16), 0, 1);

    v8f acc = {};
    acc = __builtin_amdgcn_wmma_f32_16x16x32_f16(false, a0, false, bf0,
                                                 (short)0, acc, false, false);
    acc = __builtin_amdgcn_wmma_f32_16x16x32_f16(false, a1, false, bf1,
                                                 (short)0, acc, false, false);

    const int h     = t * 16 + col;
    const float bia = b1[(size_t)n * HID + h];
    const float w2v = W2[(size_t)n * HID + h];
    const float w1c = W1[(size_t)n * KIN * HID + h];   // k=0 row (x column)
#pragma unroll
    for (int v = 0; v < 8; ++v) {
      float hv = acc[v] + xs[v] * w1c + bia;
      float e  = (hv > 0.f) ? hv : (__expf(hv) - 1.0f);  // ELU, alpha=1
      s[v] += e * w2v;
    }
  }

  // Reduce across the 16 columns held by each half-wave (lanes xor 1,2,4,8)
#pragma unroll
  for (int v = 0; v < 8; ++v) {
    s[v] += __shfl_xor(s[v], 1, 32);
    s[v] += __shfl_xor(s[v], 2, 32);
    s[v] += __shfl_xor(s[v], 4, 32);
    s[v] += __shfl_xor(s[v], 8, 32);
  }

  if (col == 0) {                                   // lanes 0 and 16 write
    const float bb = b2[n];
#pragma unroll
    for (int v = 0; v < 8; ++v)
      out[(size_t)(rbase + v) * NN + n] = s[v] + bb;
  }
}

extern "C" void kernel_launch(void* const* d_in, const int* in_sizes, int n_in,
                              void* d_out, int out_size, void* d_ws, size_t ws_size,
                              hipStream_t stream) {
  const float* x  = (const float*)d_in[0];   // (8192, 64)
  const float* z  = (const float*)d_in[1];   // (8192, 64)
  const float* W1 = (const float*)d_in[2];   // (64, 65, 256)
  const float* b1 = (const float*)d_in[3];   // (64, 256)
  const float* W2 = (const float*)d_in[4];   // (64, 256)
  const float* b2 = (const float*)d_in[5];   // (64,)
  float* out = (float*)d_out;                // (8192, 64)

  _Float16* zA  = (_Float16*)d_ws;                                // 1 MB
  _Float16* w1B = (_Float16*)((char*)d_ws + (((size_t)1) << 20)); // 2 MB

  pack_zA <<<dim3(32768 / 256), 256, 0, stream>>>(z, zA);
  pack_w1B<<<dim3(65536 / 256), 256, 0, stream>>>(W1, w1B);
  mlp_fused<<<dim3(BATCH / BMBLK, NN), WAVES * 32, 0, stream>>>(
      x, W1, b1, W2, b2, zA, w1B, out);
}
